// EitlemKKmPredictor_70334384439425
// MI455X (gfx1250) — matrix operand order repaired and checked by reading, not compiled, not run on hardware
//
#include <hip/hip_runtime.h>
#include <math.h>

// ---------------------------------------------------------------------------
// Problem constants
// ---------------------------------------------------------------------------
#define BB   128
#define LL   512
#define NN   (BB*LL)      // 65536
#define MD   1024
#define PD   1280
#define HH   128
#define NLAY 10
#define NPAT 16

typedef __bf16 bf16_t;
typedef __attribute__((ext_vector_type(16))) __bf16 v16bf;
typedef __attribute__((ext_vector_type(8)))  __bf16 v8bf;
typedef __attribute__((ext_vector_type(8)))  float  v8f;

// ---------------------------------------------------------------------------
// WMMA fragment helpers (wave32, V_WMMA_F32_16X16X32_BF16)
// A 16x32 per ISA 7.12.2: lane<16 -> M=lane, K={0..7,16..23}; lane>=16 -> K={8..15,24..31}
// B 32x16: lane<16 -> N=lane, K={0..7,16..23}; lane>=16 -> K={8..15,24..31} (same as A)
// ---------------------------------------------------------------------------
__device__ inline v16bf make_afrag(const bf16_t* __restrict__ row, int ksel) {
    v8bf lo = *(const v8bf*)(row + ksel * 8);
    v8bf hi = *(const v8bf*)(row + 16 + ksel * 8);
    v16bf a;
#pragma unroll
    for (int i = 0; i < 8; ++i) { a[i] = lo[i]; a[8 + i] = hi[i]; }
    return a;
}
__device__ inline v16bf make_afrag(const float* __restrict__ row, int ksel) {
    v8f lo = *(const v8f*)(row + ksel * 8);
    v8f hi = *(const v8f*)(row + 16 + ksel * 8);
    v16bf a;
#pragma unroll
    for (int i = 0; i < 8; ++i) { a[i] = (__bf16)lo[i]; a[8 + i] = (__bf16)hi[i]; }
    return a;
}

// ---------------------------------------------------------------------------
// Generic WMMA GEMM:  C[M,N] = act(A[M,K] @ Wt[N,K]^T + bias[N])
//   A: f32 or bf16, row-major, leading dim lda (elements)
//   Wt: bf16, N-major (pre-transposed weight), leading dim = K
//   block = 128 threads (4 waves); block tile 64(M) x 64(N); wave tile 16 x 64
//   M,N multiples of 64; K multiple of 32 (all true for this model)
//   ACT: 0 none, 1 relu, 2 exact gelu
// ---------------------------------------------------------------------------
template <typename AT, typename OT, int ACT>
__global__ __launch_bounds__(128) void gemm_wmma(
    const AT* __restrict__ A, int lda,
    const bf16_t* __restrict__ Wt,
    const float* __restrict__ bias,
    OT* __restrict__ C, int ldc, int K)
{
    const int lane = threadIdx.x & 31;
    const int wave = threadIdx.x >> 5;
    const int ksel = lane >> 4;                 // 0/1 half-wave
    const int m0   = blockIdx.y * 64 + wave * 16;
    const int n0   = blockIdx.x * 64;
    const int am   = m0 + (lane & 15);
    const int bn   = (lane & 15);

    v8f acc[4];
#pragma unroll
    for (int t = 0; t < 4; ++t)
#pragma unroll
        for (int r = 0; r < 8; ++r) acc[t][r] = 0.0f;

    const AT* arow = A + (size_t)am * lda;
    for (int k0 = 0; k0 < K; k0 += 32) {
        if (k0 + 32 < K)
            __builtin_prefetch((const void*)(arow + k0 + 32), 0, 0); // global_prefetch_b8
        v16bf afr = make_afrag(arow + k0, ksel);
#pragma unroll
        for (int t = 0; t < 4; ++t) {
            const bf16_t* bp = Wt + (size_t)(n0 + t * 16 + bn) * K + k0;
            v16bf bfr = make_afrag(bp, ksel);
            acc[t] = __builtin_amdgcn_wmma_f32_16x16x32_bf16(
                false, afr, false, bfr, (short)0, acc[t], false, false);
        }
    }

#pragma unroll
    for (int t = 0; t < 4; ++t) {
        const int col = n0 + t * 16 + bn;
        const float bv = bias ? bias[col] : 0.0f;
#pragma unroll
        for (int r = 0; r < 8; ++r) {
            const int row = m0 + r + 8 * ksel;   // C layout: vgpr r, lane half picks M/M+8
            float v = acc[t][r] + bv;
            if (ACT == 1) v = fmaxf(v, 0.0f);
            else if (ACT == 2) v = 0.5f * v * (1.0f + erff(v * 0.70710678118654752f));
            C[(size_t)row * ldc + col] = (OT)v;
        }
    }
}

// ---------------------------------------------------------------------------
// Weight prep: transpose + convert f32 -> bf16, N-major:  dst[r*K+k] = src[f(r,k)]
// ---------------------------------------------------------------------------
__global__ void prep_wt(const float* __restrict__ src, bf16_t* __restrict__ dst,
                        long total, int K, int rdiv, long s0, long s1, long s2)
{
    long i = (long)blockIdx.x * blockDim.x + threadIdx.x;
    if (i >= total) return;
    int r = (int)(i / K), k = (int)(i % K);
    dst[i] = (bf16_t)src[(long)(r / rdiv) * s0 + (long)(r % rdiv) * s1 + (long)k * s2];
}

// ---------------------------------------------------------------------------
// mol resnet: h=relu(x@w1+b1); 3x h+=relu(h@w+b); mol=relu(h). grid=B, block=128
// ---------------------------------------------------------------------------
__global__ __launch_bounds__(128) void mol_kernel(
    const float* __restrict__ x,
    const float* __restrict__ w1, const float* __restrict__ b1,
    const float* __restrict__ w2, const float* __restrict__ b2,
    const float* __restrict__ w3, const float* __restrict__ b3,
    const float* __restrict__ w4, const float* __restrict__ b4,
    float* __restrict__ mol)
{
    __shared__ float xs[MD];
    __shared__ float hb[HH];
    const int b = blockIdx.x, t = threadIdx.x;
    for (int i = t; i < MD; i += 128) xs[i] = x[(size_t)b * MD + i];
    __syncthreads();
    float acc = b1[t];
    for (int k = 0; k < MD; ++k) acc += xs[k] * w1[(size_t)k * HH + t];
    hb[t] = fmaxf(acc, 0.0f);
    __syncthreads();
    const float* ws[3] = {w2, w3, w4};
    const float* bs[3] = {b2, b3, b4};
    for (int L = 0; L < 3; ++L) {
        float v = bs[L][t];
        for (int k = 0; k < HH; ++k) v += hb[k] * ws[L][(size_t)k * HH + t];
        v = fmaxf(v, 0.0f);
        __syncthreads();
        hb[t] += v;
        __syncthreads();
    }
    mol[(size_t)b * HH + t] = fmaxf(hb[t], 0.0f);
}

// ---------------------------------------------------------------------------
// srq[b,l] = sum_o relu(mol[b]@aq_w[l] + aq_b[l])[o] * am_w[l][H+o]
// ---------------------------------------------------------------------------
__global__ __launch_bounds__(128) void srq_kernel(
    const float* __restrict__ mol, const float* __restrict__ aq_w,
    const float* __restrict__ aq_b, const float* __restrict__ am_w,
    float* __restrict__ srq)
{
    __shared__ float ms[HH];
    __shared__ float red[HH];
    const int b = blockIdx.x, l = blockIdx.y, t = threadIdx.x;
    ms[t] = mol[(size_t)b * HH + t];
    __syncthreads();
    float acc = aq_b[l * HH + t];
    const float* w = aq_w + (size_t)l * HH * HH;
    for (int h = 0; h < HH; ++h) acc += ms[h] * w[(size_t)h * HH + t];
    acc = fmaxf(acc, 0.0f);
    red[t] = acc * am_w[l * 2 * HH + HH + t];
    __syncthreads();
    for (int s = 64; s > 0; s >>= 1) { if (t < s) red[t] += red[t + s]; __syncthreads(); }
    if (t == 0) srq[b * NLAY + l] = red[0];
}

// ---------------------------------------------------------------------------
// score[n,l] = dot(kk[n, l*128:...], am_w[l, :128]) + srq[pb[n], l]
// grid=N, block=320 (10 waves, one layer per wave)
// ---------------------------------------------------------------------------
__global__ __launch_bounds__(320) void score_kernel(
    const bf16_t* __restrict__ kk, const float* __restrict__ am_w,
    const float* __restrict__ srq, const int* __restrict__ pb,
    float* __restrict__ score)
{
    const int n = blockIdx.x;
    const int l = threadIdx.x >> 5, lane = threadIdx.x & 31;
    const bf16_t* kp = kk + (size_t)n * (NLAY * HH) + l * HH + lane * 4;
    const float* aw = am_w + l * 2 * HH + lane * 4;
    float s = 0.0f;
#pragma unroll
    for (int j = 0; j < 4; ++j) s += (float)kp[j] * aw[j];
#pragma unroll
    for (int m = 16; m > 0; m >>= 1) s += __shfl_xor(s, m, 32);
    if (lane == 0) score[(size_t)n * NLAY + l] = s + srq[pb[n] * NLAY + l];
}

// ---------------------------------------------------------------------------
// segment boundaries from sorted pro_batch
// ---------------------------------------------------------------------------
__global__ void seg_bounds(const int* __restrict__ pb, int* __restrict__ start)
{
    int n = blockIdx.x * blockDim.x + threadIdx.x;
    if (n >= NN) return;
    int b = pb[n];
    int prev = (n == 0) ? -1 : pb[n - 1];
    for (int bb = prev + 1; bb <= b; ++bb) start[bb] = n;
    if (n == NN - 1) for (int bb = b + 1; bb <= BB; ++bb) start[bb] = NN;
}

// segment max + sum(exp). grid=(B,NL), block=256
__global__ __launch_bounds__(256) void seg_stats(
    const float* __restrict__ score, const int* __restrict__ start,
    float* __restrict__ mbuf, float* __restrict__ sbuf)
{
    __shared__ float red[256];
    const int b = blockIdx.x, l = blockIdx.y, t = threadIdx.x;
    const int s0 = start[b], s1 = start[b + 1];
    float m = -INFINITY;
    for (int i = s0 + t; i < s1; i += 256) m = fmaxf(m, score[(size_t)i * NLAY + l]);
    red[t] = m; __syncthreads();
    for (int s = 128; s > 0; s >>= 1) { if (t < s) red[t] = fmaxf(red[t], red[t + s]); __syncthreads(); }
    m = red[0]; __syncthreads();
    float acc = 0.0f;
    for (int i = s0 + t; i < s1; i += 256) acc += __expf(score[(size_t)i * NLAY + l] - m);
    red[t] = acc; __syncthreads();
    for (int s = 128; s > 0; s >>= 1) { if (t < s) red[t] += red[t + s]; __syncthreads(); }
    if (t == 0) { mbuf[b * NLAY + l] = m; sbuf[b * NLAY + l] = red[0]; }
}

// att_pro[b,l,h] = sum_{n in seg(b)} kk[n,l,h] * w[n,l]. grid=(B,NL), block=128
__global__ __launch_bounds__(128) void att_pro_kernel(
    const bf16_t* __restrict__ kk, const float* __restrict__ score,
    const int* __restrict__ start, const float* __restrict__ mbuf,
    const float* __restrict__ sbuf, float* __restrict__ att_pro)
{
    __shared__ float wls[128];
    const int b = blockIdx.x, l = blockIdx.y, h = threadIdx.x;
    const int s0 = start[b], s1 = start[b + 1];
    const float m = mbuf[b * NLAY + l];
    const float inv = 1.0f / (sbuf[b * NLAY + l] + 1e-16f);
    float acc = 0.0f;
    for (int base = s0; base < s1; base += 128) {
        const int c = min(128, s1 - base);
        __syncthreads();
        if (h < c) wls[h] = __expf(score[(size_t)(base + h) * NLAY + l] - m) * inv;
        __syncthreads();
        for (int j = 0; j < c; ++j)
            acc += (float)kk[(size_t)(base + j) * (NLAY * HH) + l * HH + h] * wls[j];
    }
    att_pro[(size_t)(b * NLAY + l) * HH + h] = acc;
}

// ---------------------------------------------------------------------------
// LayerNorm over 1024 of x, two parameter sets -> bf16 outputs (GEMM inputs)
// ---------------------------------------------------------------------------
__global__ __launch_bounds__(256) void ln1024_kernel(
    const float* __restrict__ x,
    const float* __restrict__ g1, const float* __restrict__ b1,
    const float* __restrict__ g2, const float* __restrict__ b2,
    bf16_t* __restrict__ o1, bf16_t* __restrict__ o2)
{
    __shared__ float red[256];
    const int b = blockIdx.x, t = threadIdx.x;
    float s = 0.0f, ss = 0.0f;
    for (int i = t; i < MD; i += 256) { float v = x[(size_t)b * MD + i]; s += v; ss += v * v; }
    red[t] = s; __syncthreads();
    for (int k = 128; k > 0; k >>= 1) { if (t < k) red[t] += red[t + k]; __syncthreads(); }
    const float mean = red[0] / (float)MD; __syncthreads();
    red[t] = ss; __syncthreads();
    for (int k = 128; k > 0; k >>= 1) { if (t < k) red[t] += red[t + k]; __syncthreads(); }
    const float var = red[0] / (float)MD - mean * mean;
    const float rstd = rsqrtf(var + 1e-5f);
    for (int i = t; i < MD; i += 256) {
        const float v = (x[(size_t)b * MD + i] - mean) * rstd;
        o1[(size_t)b * MD + i] = (bf16_t)(v * g1[i] + b1[i]);
        o2[(size_t)b * MD + i] = (bf16_t)(v * g2[i] + b2[i]);
    }
}

// LayerNorm over 128. grid = rows, block = 128
__global__ __launch_bounds__(128) void ln128_kernel(
    const float* __restrict__ in, const float* __restrict__ g,
    const float* __restrict__ bb, float* __restrict__ out)
{
    __shared__ float red[128];
    const int r = blockIdx.x, t = threadIdx.x;
    const float v = in[(size_t)r * HH + t];
    red[t] = v; __syncthreads();
    for (int k = 64; k > 0; k >>= 1) { if (t < k) red[t] += red[t + k]; __syncthreads(); }
    const float mean = red[0] / (float)HH; __syncthreads();
    const float d = v - mean;
    red[t] = d * d; __syncthreads();
    for (int k = 64; k > 0; k >>= 1) { if (t < k) red[t] += red[t + k]; __syncthreads(); }
    const float rstd = rsqrtf(red[0] / (float)HH + 1e-5f);
    out[(size_t)r * HH + t] = d * rstd * g[t] + bb[t];
}

// ---------------------------------------------------------------------------
// Small FC: out[r,o] = act(dot(concat(in1[r],in2[r]), W[:,o]) + bias[o])
// grid = rows, block = N, dyn-shared = (K1+K2)*4. ACT: 0 none, 2 gelu
// ---------------------------------------------------------------------------
template <int ACT>
__global__ __launch_bounds__(128) void small_fc(
    const float* __restrict__ in1, int K1, const float* __restrict__ in2, int K2,
    const float* __restrict__ W, const float* __restrict__ bias,
    float* __restrict__ out, int Ncol)
{
    extern __shared__ float sh[];
    const int r = blockIdx.x, o = threadIdx.x;
    const int K = K1 + K2;
    for (int i = threadIdx.x; i < K1; i += blockDim.x) sh[i] = in1[(size_t)r * K1 + i];
    if (in2) for (int i = threadIdx.x; i < K2; i += blockDim.x) sh[K1 + i] = in2[(size_t)r * K2 + i];
    __syncthreads();
    float acc = bias ? bias[o] : 0.0f;
    for (int k = 0; k < K; ++k) acc += sh[k] * W[(size_t)k * Ncol + o];
    if (ACT == 2) acc = 0.5f * acc * (1.0f + erff(acc * 0.70710678118654752f));
    out[(size_t)r * Ncol + o] = acc;
}

// mean over NL layers of att_pro
__global__ __launch_bounds__(128) void apmean_kernel(const float* __restrict__ att_pro,
                                                     float* __restrict__ out)
{
    const int b = blockIdx.x, h = threadIdx.x;
    float s = 0.0f;
    for (int l = 0; l < NLAY; ++l) s += att_pro[(size_t)(b * NLAY + l) * HH + h];
    out[(size_t)b * HH + h] = s * (1.0f / (float)NLAY);
}

// attention pool: att=softmax(q.K/sqrt(H)) over T; out = att @ V. grid=B, block=128
__global__ __launch_bounds__(128) void pool_kernel(
    const float* __restrict__ q_all, const float* __restrict__ Kb,
    const float* __restrict__ Vb, int T, float* __restrict__ out)
{
    __shared__ float qrow[HH];
    __shared__ float att[NPAT];
    const int b = blockIdx.x, t = threadIdx.x;
    qrow[t] = q_all[(size_t)b * HH + t];
    __syncthreads();
    if (t < T) {
        const float* kr = Kb + ((size_t)b * T + t) * HH;
        float s = 0.0f;
        for (int h = 0; h < HH; ++h) s += qrow[h] * kr[h];
        att[t] = s * 0.08838834764831845f;  // 1/sqrt(128)
    }
    __syncthreads();
    if (t == 0) {
        float m = -INFINITY;
        for (int i = 0; i < T; ++i) m = fmaxf(m, att[i]);
        float sum = 0.0f;
        for (int i = 0; i < T; ++i) { att[i] = __expf(att[i] - m); sum += att[i]; }
        const float inv = 1.0f / sum;
        for (int i = 0; i < T; ++i) att[i] *= inv;
    }
    __syncthreads();
    float acc = 0.0f;
    for (int i = 0; i < T; ++i) acc += att[i] * Vb[((size_t)b * T + i) * HH + t];
    out[(size_t)b * HH + t] = acc;
}

// final MLP: prelu(o1) -> prelu(o2) -> o3. grid=B, block=512
__global__ __launch_bounds__(512) void tail_out_kernel(
    const float* __restrict__ vd, const float* __restrict__ vp,
    const float* __restrict__ o1w, const float* __restrict__ o1b, const float* __restrict__ pr1,
    const float* __restrict__ o2w, const float* __restrict__ o2b, const float* __restrict__ pr2,
    const float* __restrict__ o3w, const float* __restrict__ o3b,
    float* __restrict__ out)
{
    __shared__ float h0[2 * HH];
    __shared__ float h1[4 * HH];
    __shared__ float h2[HH];
    const int b = blockIdx.x, t = threadIdx.x;
    if (t < HH) h0[t] = vd[(size_t)b * HH + t];
    else if (t < 2 * HH) h0[t] = vp[(size_t)b * HH + (t - HH)];
    __syncthreads();
    {
        const float a = *pr1;
        float acc = o1b[t];
        for (int k = 0; k < 2 * HH; ++k) acc += h0[k] * o1w[(size_t)k * 4 * HH + t];
        h1[t] = acc > 0.0f ? acc : a * acc;
    }
    __syncthreads();
    if (t < HH) {
        const float a = *pr2;
        float acc = o2b[t];
        for (int k = 0; k < 4 * HH; ++k) acc += h1[k] * o2w[(size_t)k * HH + t];
        h2[t] = acc > 0.0f ? acc : a * acc;
    }
    __syncthreads();
    if (t == 0) {
        float acc = o3b[0];
        for (int k = 0; k < HH; ++k) acc += h2[k] * o3w[k];
        out[b] = acc;
    }
}

// ---------------------------------------------------------------------------
// Launch
// ---------------------------------------------------------------------------
extern "C" void kernel_launch(void* const* d_in, const int* in_sizes, int n_in,
                              void* d_out, int out_size, void* d_ws, size_t ws_size,
                              hipStream_t stream)
{
    (void)in_sizes; (void)n_in; (void)out_size; (void)ws_size;
    const float* x        = (const float*)d_in[0];
    const float* pro_emb  = (const float*)d_in[1];
    const int*   pb       = (const int*)  d_in[2];
    const float* r_w1 = (const float*)d_in[3],  *r_b1 = (const float*)d_in[4];
    const float* r_w2 = (const float*)d_in[5],  *r_b2 = (const float*)d_in[6];
    const float* r_w3 = (const float*)d_in[7],  *r_b3 = (const float*)d_in[8];
    const float* r_w4 = (const float*)d_in[9],  *r_b4 = (const float*)d_in[10];
    const float* prej2_w = (const float*)d_in[11];
    const float* aq_w = (const float*)d_in[12], *aq_b = (const float*)d_in[13];
    const float* ak_w = (const float*)d_in[14], *ak_b = (const float*)d_in[15];
    const float* am_w = (const float*)d_in[16];
    const float* fpln_g = (const float*)d_in[17], *fpln_b = (const float*)d_in[18];
    const float* fpproj_w = (const float*)d_in[19], *fpproj_b = (const float*)d_in[20];
    const float* gdln_g = (const float*)d_in[21], *gdln_b = (const float*)d_in[22];
    const float* gd1_w = (const float*)d_in[23], *gd1_b = (const float*)d_in[24];
    const float* gd2_w = (const float*)d_in[25], *gd2_b = (const float*)d_in[26];
    const float* gp_w  = (const float*)d_in[27], *gp_b  = (const float*)d_in[28];
    const float* q_w   = (const float*)d_in[29], *q_b   = (const float*)d_in[30];
    const float* lnd_g = (const float*)d_in[31], *lnd_b = (const float*)d_in[32];
    const float* kd_w  = (const float*)d_in[33], *kd_b  = (const float*)d_in[34];
    const float* lnp_g = (const float*)d_in[35], *lnp_b = (const float*)d_in[36];
    const float* kp_w  = (const float*)d_in[37], *kp_b  = (const float*)d_in[38];
    const float* o1_w  = (const float*)d_in[39], *o1_b  = (const float*)d_in[40];
    const float* pr1   = (const float*)d_in[41];
    const float* o2_w  = (const float*)d_in[42], *o2_b  = (const float*)d_in[43];
    const float* pr2   = (const float*)d_in[44];
    const float* o3_w  = (const float*)d_in[45], *o3_b  = (const float*)d_in[46];
    float* out = (float*)d_out;

    // workspace carve-up (256B aligned)
    char* base = (char*)d_ws;
    size_t off = 0;
    auto alloc = [&](size_t bytes) -> void* {
        void* p = base + off;
        off += (bytes + 255) & ~(size_t)255;
        return p;
    };
    bf16_t* WT1   = (bf16_t*)alloc((size_t)HH * PD * 2);            // prej2_w^T
    bf16_t* WT2   = (bf16_t*)alloc((size_t)NLAY * HH * HH * 2);     // ak_w^T (per layer)
    bf16_t* WTfp  = (bf16_t*)alloc((size_t)NPAT * HH * MD * 2);     // fpproj_w^T
    bf16_t* WTgd  = (bf16_t*)alloc((size_t)HH * MD * 2);            // gd1_w^T
    bf16_t* prot  = (bf16_t*)alloc((size_t)NN * HH * 2);
    bf16_t* kk    = (bf16_t*)alloc((size_t)NN * NLAY * HH * 2);     // 168 MB
    float*  score = (float*) alloc((size_t)NN * NLAY * 4);
    float*  srq   = (float*) alloc((size_t)BB * NLAY * 4);
    float*  mol   = (float*) alloc((size_t)BB * HH * 4);
    int*    segs  = (int*)   alloc((size_t)(BB + 1) * 4);
    float*  mbuf  = (float*) alloc((size_t)BB * NLAY * 4);
    float*  sbuf  = (float*) alloc((size_t)BB * NLAY * 4);
    float*  attp  = (float*) alloc((size_t)BB * NLAY * HH * 4);
    bf16_t* lnfp  = (bf16_t*)alloc((size_t)BB * MD * 2);
    bf16_t* lnfp2 = (bf16_t*)alloc((size_t)BB * MD * 2);
    float*  draw  = (float*) alloc((size_t)BB * NPAT * HH * 4);
    float*  dtok  = (float*) alloc((size_t)BB * NPAT * HH * 4);
    float*  Kd    = (float*) alloc((size_t)BB * NPAT * HH * 4);
    float*  apm   = (float*) alloc((size_t)BB * HH * 4);
    float*  g1    = (float*) alloc((size_t)BB * HH * 4);
    float*  gd    = (float*) alloc((size_t)BB * HH * 4);
    float*  gp    = (float*) alloc((size_t)BB * HH * 4);
    float*  qall  = (float*) alloc((size_t)BB * HH * 4);
    float*  Pbuf  = (float*) alloc((size_t)BB * NLAY * HH * 4);
    float*  Kp    = (float*) alloc((size_t)BB * NLAY * HH * 4);
    float*  vd    = (float*) alloc((size_t)BB * HH * 4);
    float*  vp    = (float*) alloc((size_t)BB * HH * 4);

    // --- weight prep (bf16, N-major) ---
    {
        long tot;
        tot = (long)HH * PD;                 // WT1[o][k] = prej2_w[k][o]
        prep_wt<<<(tot + 255) / 256, 256, 0, stream>>>(prej2_w, WT1, tot, PD, HH, 0, 1, HH);
        tot = (long)NLAY * HH * HH;          // WT2[l*H+o][k] = ak_w[l][k][o]
        prep_wt<<<(tot + 255) / 256, 256, 0, stream>>>(ak_w, WT2, tot, HH, HH, (long)HH * HH, 1, HH);
        tot = (long)NPAT * HH * MD;          // WTfp[o][k] = fpproj_w[k][o]
        prep_wt<<<(tot + 255) / 256, 256, 0, stream>>>(fpproj_w, WTfp, tot, MD, NPAT * HH, 0, 1, NPAT * HH);
        tot = (long)HH * MD;                 // WTgd[o][k] = gd1_w[k][o]
        prep_wt<<<(tot + 255) / 256, 256, 0, stream>>>(gd1_w, WTgd, tot, MD, HH, 0, 1, HH);
    }

    // --- small front: mol resnet, srq, LN(x) x2, segment bounds ---
    mol_kernel<<<BB, 128, 0, stream>>>(x, r_w1, r_b1, r_w2, r_b2, r_w3, r_b3, r_w4, r_b4, mol);
    srq_kernel<<<dim3(BB, NLAY), 128, 0, stream>>>(mol, aq_w, aq_b, am_w, srq);
    ln1024_kernel<<<BB, 256, 0, stream>>>(x, fpln_g, fpln_b, gdln_g, gdln_b, lnfp, lnfp2);
    seg_bounds<<<(NN + 255) / 256, 256, 0, stream>>>(pb, segs);

    // --- big WMMA GEMMs ---
    // prot = relu(pro_emb @ prej2_w)   [65536 x 128], K=1280
    gemm_wmma<float, bf16_t, 1><<<dim3(HH / 64, NN / 64), 128, 0, stream>>>(
        pro_emb, PD, WT1, nullptr, prot, HH, PD);
    // kk = relu(prot @ ak_w + ak_b)    [65536 x 1280], K=128
    gemm_wmma<bf16_t, bf16_t, 1><<<dim3(NLAY * HH / 64, NN / 64), 128, 0, stream>>>(
        prot, HH, WT2, ak_b, kk, NLAY * HH, HH);

    // --- segmented attention over proteins ---
    score_kernel<<<NN, NLAY * 32, 0, stream>>>(kk, am_w, srq, pb, score);
    seg_stats<<<dim3(BB, NLAY), 256, 0, stream>>>(score, segs, mbuf, sbuf);
    att_pro_kernel<<<dim3(BB, NLAY), 128, 0, stream>>>(kk, score, segs, mbuf, sbuf, attp);

    // --- fp branch (WMMA for the two K=1024 GEMMs) ---
    // d_raw = LN(x) @ fpproj_w + b   [128 x 2048]
    gemm_wmma<bf16_t, float, 0><<<dim3(NPAT * HH / 64, BB / 64), 128, 0, stream>>>(
        lnfp, MD, WTfp, fpproj_b, draw, NPAT * HH, MD);
    // g1 = gelu(LN(x) @ gd1_w + b)   [128 x 128]
    gemm_wmma<bf16_t, float, 2><<<dim3(HH / 64, BB / 64), 128, 0, stream>>>(
        lnfp2, MD, WTgd, gd1_b, g1, HH, MD);

    ln128_kernel<<<BB * NPAT, 128, 0, stream>>>(draw, lnd_g, lnd_b, dtok);
    small_fc<0><<<BB * NPAT, 128, HH * 4, stream>>>(dtok, HH, nullptr, 0, kd_w, kd_b, Kd, HH);
    small_fc<0><<<BB, 128, HH * 4, stream>>>(g1, HH, nullptr, 0, gd2_w, gd2_b, gd, HH);

    apmean_kernel<<<BB, 128, 0, stream>>>(attp, apm);
    small_fc<2><<<BB, 128, HH * 4, stream>>>(apm, HH, nullptr, 0, gp_w, gp_b, gp, HH);
    small_fc<0><<<BB, 128, 2 * HH * 4, stream>>>(gd, HH, gp, HH, q_w, q_b, qall, HH);

    ln128_kernel<<<BB * NLAY, 128, 0, stream>>>(attp, lnp_g, lnp_b, Pbuf);
    small_fc<0><<<BB * NLAY, 128, HH * 4, stream>>>(Pbuf, HH, nullptr, 0, kp_w, kp_b, Kp, HH);

    pool_kernel<<<BB, 128, 0, stream>>>(qall, Kp, Pbuf, NLAY, vp);
    pool_kernel<<<BB, 128, 0, stream>>>(qall, Kd, dtok, NPAT, vd);

    tail_out_kernel<<<BB, 512, 0, stream>>>(vd, vp, o1_w, o1_b, pr1, o2_w, o2_b, pr2, o3_w, o3_b, out);
}
